// PCSA_66597762892576
// MI455X (gfx1250) — matrix operand order repaired and checked by reading, not compiled, not run on hardware
//
#include <hip/hip_runtime.h>
#include <cstdint>

// ---------------------------------------------------------------------------
// Point-cloud spectral attention for MI455X (gfx1250, wave32, WMMA).
//   - Spectral einsum (68.7 GFLOP) -> v_wmma_f32_16x16x32_bf16 (f32 accum),
//     b128 LDS fragment loads, b128 gather/copy tile builds.
//   - SE + fusion GEMMs (10.7 GFLOP) -> v_wmma_f32_16x16x4_f32 (full fp32)
//   - KNN top-16, BatchNorm: deterministic (no float atomics)
// ---------------------------------------------------------------------------

typedef __attribute__((ext_vector_type(16))) __bf16   v16bf;
typedef __attribute__((ext_vector_type(8)))  float    v8f;
typedef __attribute__((ext_vector_type(4)))  float    v4f;
typedef __attribute__((ext_vector_type(2)))  float    v2f;
typedef __attribute__((ext_vector_type(4)))  uint32_t v4u;

union BFrag { v4f f[2]; v16bf v; };   // 32B: two b128 LDS loads -> one fragment
union Pk4   { __bf16 h[4]; uint64_t u; };

constexpr int Bsz = 8, Cc = 256, Nn = 4096, Kk = 16;
constexpr int Mpts = Bsz * Nn;       // 32768 points
constexpr float BN_EPS = 1e-5f;

// ------------------------- transpose x (B,C,N) -> x_t (B,N,C) --------------
__global__ void k_transpose_xt(const float* __restrict__ x, float* __restrict__ xt) {
  __shared__ float tile[32][33];
  int b = blockIdx.z;
  int n0 = blockIdx.x * 32, c0 = blockIdx.y * 32;
  int tx = threadIdx.x, ty = threadIdx.y;            // block (32, 8)
#pragma unroll
  for (int i = 0; i < 4; ++i)
    tile[ty + i * 8][tx] = x[(size_t)b * Cc * Nn + (size_t)(c0 + ty + i * 8) * Nn + n0 + tx];
  __syncthreads();
#pragma unroll
  for (int i = 0; i < 4; ++i)
    xt[(size_t)b * Nn * Cc + (size_t)(n0 + ty + i * 8) * Cc + c0 + tx] = tile[tx][ty + i * 8];
}

// ---------- filter (KK=4096, D=256) f32 -> transposed bf16 (D, KK) ---------
__global__ void k_filter_bf16T(const float* __restrict__ f, __bf16* __restrict__ fbT) {
  __shared__ float tile[32][33];
  int kk0 = blockIdx.x * 32, d0 = blockIdx.y * 32;
  int tx = threadIdx.x, ty = threadIdx.y;            // block (32, 8)
#pragma unroll
  for (int i = 0; i < 4; ++i)
    tile[ty + i * 8][tx] = f[(size_t)(kk0 + ty + i * 8) * 256 + d0 + tx];
  __syncthreads();
#pragma unroll
  for (int i = 0; i < 4; ++i)
    fbT[(size_t)(d0 + ty + i * 8) * 4096 + kk0 + tx] = (__bf16)tile[tx][ty + i * 8];
}

// ------------------------- KNN: top-16 smallest d2 per query ---------------
__global__ void k_knn(const float* __restrict__ pos, int* __restrict__ idx) {
  __shared__ float spx[256], spy[256], spz[256];
  int t = threadIdx.x;
  int q = blockIdx.x * 256 + t;                 // global point id
  int b = q >> 12;                              // / 4096 (block never straddles b)
  int n = q & (Nn - 1);
  const float* pb = pos + (size_t)b * 3 * Nn;
  float px = pb[n], py = pb[Nn + n], pz = pb[2 * Nn + n];
  float sqq = px * px + py * py + pz * pz;
  float bd[16]; int bi[16];
#pragma unroll
  for (int i = 0; i < 16; ++i) { bd[i] = 3.4e38f; bi[i] = 0; }
  for (int j0 = 0; j0 < Nn; j0 += 256) {
    __syncthreads();
    spx[t] = pb[j0 + t];
    spy[t] = pb[Nn + j0 + t];
    spz[t] = pb[2 * Nn + j0 + t];
    __syncthreads();
    for (int jj = 0; jj < 256; ++jj) {
      float qx = spx[jj], qy = spy[jj], qz = spz[jj];
      float sqj = qx * qx + qy * qy + qz * qz;
      float inner = px * qx + py * qy + pz * qz;
      float d2 = sqq + sqj - 2.0f * inner;      // same formula as reference
      int j = j0 + jj;
      if (d2 < bd[15]) {                        // sorted insert; strict < keeps
#pragma unroll                                  // lower index on ties (top_k)
        for (int i = 15; i >= 1; --i) {
          if (d2 < bd[i - 1])      { bd[i] = bd[i - 1]; bi[i] = bi[i - 1]; }
          else if (d2 < bd[i])     { bd[i] = d2;        bi[i] = j;         }
        }
        if (d2 < bd[0]) { bd[0] = d2; bi[0] = j; }
      }
    }
  }
#pragma unroll
  for (int i = 0; i < 16; ++i) idx[(size_t)q * 16 + i] = bi[i];
}

// ------------------------- spectral GEMM: bf16 WMMA ------------------------
// s[m,d] = sum_{k,c} (x_t[nbr(m,k),c] - x_t[m,c]) * filter[k,c,d]
// A tile (gathered edge, 64x32) row-major bf16; B tile (128 d x 32 k) d-major
// bf16 from pre-transposed filter; strides 40 bf16 (80B rows: 16B aligned for
// b128 fragment loads, bank step 20 -> conflict-free). Center rows cached in
// LDS per c-slab; neighbor ids preloaded once. 2 barriers per k-step.
constexpr int G1_AS = 40;
constexpr int G1_BS = 40;
constexpr int G1_CS = 36;   // f32 center tile stride (144B rows, 16B aligned)

__global__ void k_spectral_gemm(const float* __restrict__ xt,
                                const int* __restrict__ idx,
                                const __bf16* __restrict__ fbT,
                                float* __restrict__ s) {
  __shared__ __bf16 As[64 * G1_AS];
  __shared__ __bf16 Bs[128 * G1_BS];
  __shared__ float  Ctr[64 * G1_CS];
  __shared__ int    nb16[64 * 16];
  int t = threadIdx.x;
  int lane = t & 31, wave = t >> 5;
  int wm = wave >> 1, wd = wave & 1;             // 4x2 wave grid, 16x64 per wave
  int m0 = blockIdx.x * 64, d0 = blockIdx.y * 128;
  int bBase = (m0 >> 12) << 12;                  // b * Nn
  int mL = lane & 15, hi = lane >> 4;

  // preload all 64x16 neighbor ids once
#pragma unroll
  for (int i = 0; i < 4; ++i) {
    int e = t + i * 256;
    nb16[e] = idx[(size_t)(m0 + (e >> 4)) * 16 + (e & 15)];
  }

  v8f acc[4] = {};
  for (int c0 = 0; c0 < 256; c0 += 32) {         // c-slab outer: cache centers
    // center tile 64x32 f32 (b128 copies); consumed after next barrier
#pragma unroll
    for (int i = 0; i < 2; ++i) {
      int e = t + i * 256;
      int mi = e >> 3, c4 = (e & 7) * 4;
      *(v4f*)&Ctr[mi * G1_CS + c4] = *(const v4f*)&xt[(size_t)(m0 + mi) * 256 + c0 + c4];
    }
    for (int k = 0; k < Kk; ++k) {
      int kk0 = k * 256 + c0;
      __syncthreads();                           // prev frag reads done
      // B tile: one b128 load + one b128 LDS store per thread
      {
        int dj = t >> 1, c8 = (t & 1) * 16;
        *(v4u*)&Bs[dj * G1_BS + c8] =
            *(const v4u*)&fbT[(size_t)(d0 + dj) * 4096 + kk0 + c8];
      }
      // A tile: b128 gather, f32 subtract, pack 4 bf16, b64 store
#pragma unroll
      for (int i = 0; i < 2; ++i) {
        int e = t + i * 256;
        int mi = e >> 3, c4 = (e & 7) * 4;
        v4f nv = *(const v4f*)&xt[(size_t)(bBase + nb16[mi * 16 + k]) * 256 + c0 + c4];
        v4f cv = *(const v4f*)&Ctr[mi * G1_CS + c4];
        Pk4 pk;
#pragma unroll
        for (int l = 0; l < 4; ++l) pk.h[l] = (__bf16)(nv[l] - cv[l]);
        *(uint64_t*)&As[mi * G1_AS + c4] = pk.u;
      }
      __syncthreads();
      // A fragment: two contiguous 8-elem runs -> 2x ds_load_b128
      BFrag ua;
      int arow = (wm * 16 + mL) * G1_AS + hi * 8;
      ua.f[0] = *(const v4f*)&As[arow];
      ua.f[1] = *(const v4f*)&As[arow + 16];
      // hoist all B fragments, then back-to-back WMMAs
      BFrag ub[4];
#pragma unroll
      for (int j = 0; j < 4; ++j) {
        int brow = (wd * 64 + j * 16 + mL) * G1_BS + hi * 8;
        ub[j].f[0] = *(const v4f*)&Bs[brow];
        ub[j].f[1] = *(const v4f*)&Bs[brow + 16];
      }
#pragma unroll
      for (int j = 0; j < 4; ++j)
        acc[j] = __builtin_amdgcn_wmma_f32_16x16x32_bf16(
            false, ua.v, false, ub[j].v, (short)0, acc[j], false, false);
    }
  }
  // C/D layout: lanes 0-15 -> M=r, lanes 16-31 -> M=8+r; N = lane&15
#pragma unroll
  for (int j = 0; j < 4; ++j)
#pragma unroll
    for (int r = 0; r < 8; ++r)
      s[(size_t)(m0 + wm * 16 + r + hi * 8) * 256 + d0 + wd * 64 + j * 16 + mL] = acc[j][r];
}

// ------------------------- BatchNorm (deterministic two-stage) -------------
__global__ void k_bn_partial(const float* __restrict__ s,
                             float* __restrict__ psum, float* __restrict__ psq) {
  int d = threadIdx.x;
  int r0 = blockIdx.x * 128;
  float sm = 0.f, sq = 0.f;
  for (int r = 0; r < 128; ++r) {
    float v = s[(size_t)(r0 + r) * 256 + d];
    sm += v; sq += v * v;
  }
  psum[blockIdx.x * 256 + d] = sm;
  psq [blockIdx.x * 256 + d] = sq;
}

__global__ void k_bn_final(const float* __restrict__ psum, const float* __restrict__ psq,
                           const float* __restrict__ gamma, const float* __restrict__ beta,
                           float* __restrict__ scale, float* __restrict__ shift) {
  int d = threadIdx.x;
  float sm = 0.f, sq = 0.f;
  for (int p = 0; p < Mpts / 128; ++p) { sm += psum[p * 256 + d]; sq += psq[p * 256 + d]; }
  float inv = 1.f / (float)Mpts;
  float mean = sm * inv;
  float var  = sq * inv - mean * mean;
  float sc = gamma[d] * rsqrtf(var + BN_EPS);
  scale[d] = sc;
  shift[d] = beta[d] - mean * sc;
}

__global__ void k_bn_apply(float* __restrict__ s, const float* __restrict__ scale,
                           const float* __restrict__ shift) {
  int i = blockIdx.x * 256 + threadIdx.x;
  int d = i & 255;
  s[i] = s[i] * scale[d] + shift[d];
}

// ------------------------- generic fp32 WMMA GEMM --------------------------
// out(M,N) = act( A(M,KK) * B^T + bias )   with B stored as weights[n*ldb+kk]
// A is split: kk <  splitK -> A0,  kk >= splitK -> A1 (concat along KK; the
// switch is uniform per 32-wide k-step since splitK % 32 == 0).
// act: 0 = bias only; 1 = relu+bias; 2 = sigmoid+bias then * mul[m*ldo+dcol]
// outmode: 0 row-major out[m*ldo+dcol]; 1 transposed (B,C,N) store for d_out.
constexpr int G2_AS = 36, G2_BS = 36;  // 144B rows: 16B aligned tile copies

__global__ void k_gemm_f32(const float* __restrict__ A0, int lda0,
                           const float* __restrict__ A1, int lda1, int splitK,
                           const float* __restrict__ Bw, int ldb,
                           const float* __restrict__ bias,
                           int KK, int act,
                           const float* __restrict__ mul,
                           float* __restrict__ out, int ldo, int outmode) {
  __shared__ float As [64 * G2_AS];
  __shared__ float BsT[64 * G2_BS];              // n-major: BsT[n][k]
  int t = threadIdx.x;
  int lane = t & 31, wave = t >> 5;
  int wm = wave >> 1, wd = wave & 1;             // 4x2 wave grid, 16x32 per wave
  int m0 = blockIdx.x * 64, n0 = blockIdx.y * 64;
  int nL = lane & 15, hi = lane >> 4;
  v8f acc[2] = {};

  for (int kk0 = 0; kk0 < KK; kk0 += 32) {
    const float* Asrc; int lda;
    if (kk0 < splitK) { Asrc = A0 + kk0;            lda = lda0; }
    else              { Asrc = A1 + (kk0 - splitK); lda = lda1; }
    __syncthreads();
#pragma unroll
    for (int i = 0; i < 2; ++i) {                // A tile 64x32, b128 copies
      int e = t + i * 256;
      int mi = e >> 3, c4 = (e & 7) * 4;
      *(v4f*)&As[mi * G2_AS + c4] = *(const v4f*)&Asrc[(size_t)(m0 + mi) * lda + c4];
    }
#pragma unroll
    for (int i = 0; i < 2; ++i) {                // B tile 64n x 32k, b128
      int e = t + i * 256;
      int ni = e >> 3, c4 = (e & 7) * 4;
      *(v4f*)&BsT[ni * G2_BS + c4] = *(const v4f*)&Bw[(size_t)(n0 + ni) * ldb + kk0 + c4];
    }
    __syncthreads();
#pragma unroll
    for (int kt = 0; kt < 8; ++kt) {
      // A 16x4 frag: VGPR0 K=0|K=2, VGPR1 K=1|K=3 (low|high lanes): b64
      v2f a = *(const v2f*)&As[(wm * 16 + nL) * G2_AS + kt * 4 + 2 * hi];
#pragma unroll
      for (int j = 0; j < 2; ++j) {
        // B 4x16 frag: low lanes K=0,1; high lanes K=2,3: b64
        v2f b = *(const v2f*)&BsT[(wd * 32 + j * 16 + nL) * G2_BS + kt * 4 + 2 * hi];
        acc[j] = __builtin_amdgcn_wmma_f32_16x16x4_f32(
            false, a, false, b, (short)0, acc[j], false, false);
      }
    }
  }
#pragma unroll
  for (int j = 0; j < 2; ++j) {
    int dcol = n0 + wd * 32 + j * 16 + nL;
    float bv = bias[dcol];
#pragma unroll
    for (int r = 0; r < 8; ++r) {
      int m = m0 + wm * 16 + r + hi * 8;
      float v = acc[j][r] + bv;
      if (act == 1)      v = v > 0.f ? v : 0.f;
      else if (act == 2) { v = 1.f / (1.f + __expf(-v)); v *= mul[(size_t)m * ldo + dcol]; }
      if (outmode == 0)  out[(size_t)m * ldo + dcol] = v;
      else               out[((size_t)(m >> 12) << 20) + (size_t)dcol * Nn + (m & (Nn - 1))] = v;
    }
  }
}

// ---------------------------------------------------------------------------
extern "C" void kernel_launch(void* const* d_in, const int* in_sizes, int n_in,
                              void* d_out, int out_size, void* d_ws, size_t ws_size,
                              hipStream_t stream) {
  (void)in_sizes; (void)n_in; (void)out_size; (void)ws_size;
  const float* x     = (const float*)d_in[0];
  const float* pos   = (const float*)d_in[1];
  const float* filt  = (const float*)d_in[2];
  const float* gamma = (const float*)d_in[3];
  const float* beta  = (const float*)d_in[4];
  const float* w1    = (const float*)d_in[5];
  const float* b1    = (const float*)d_in[6];
  const float* w2    = (const float*)d_in[7];
  const float* b2    = (const float*)d_in[8];
  const float* wf    = (const float*)d_in[9];
  const float* bf    = (const float*)d_in[10];
  float* out = (float*)d_out;

  char* w = (char*)d_ws;
  size_t off = 0;
  float* xt   = (float*)(w + off); off += (size_t)Mpts * Cc * 4;        // 33.5 MB
  float* s    = (float*)(w + off); off += (size_t)Mpts * Cc * 4;        // 33.5 MB
  float* h    = (float*)(w + off); off += (size_t)Mpts * 64 * 4;        //  8.4 MB
  int*   idx  = (int*)  (w + off); off += (size_t)Mpts * Kk * 4;        //  2.1 MB
  __bf16* fbT = (__bf16*)(w + off); off += (size_t)Kk * Cc * Cc * 2;    //  2.1 MB
  float* psum = (float*)(w + off); off += (size_t)(Mpts / 128) * 256 * 4;
  float* psq  = (float*)(w + off); off += (size_t)(Mpts / 128) * 256 * 4;
  float* bnsc = (float*)(w + off); off += 256 * 4;
  float* bnsh = (float*)(w + off); off += 256 * 4;

  k_transpose_xt<<<dim3(Nn / 32, Cc / 32, Bsz), dim3(32, 8), 0, stream>>>(x, xt);
  k_filter_bf16T<<<dim3((Kk * Cc) / 32, Cc / 32), dim3(32, 8), 0, stream>>>(filt, fbT);
  k_knn<<<Mpts / 256, 256, 0, stream>>>(pos, idx);
  k_spectral_gemm<<<dim3(Mpts / 64, Cc / 128), 256, 0, stream>>>(xt, idx, fbT, s);
  k_bn_partial<<<Mpts / 128, 256, 0, stream>>>(s, psum, psq);
  k_bn_final<<<1, 256, 0, stream>>>(psum, psq, gamma, beta, bnsc, bnsh);
  k_bn_apply<<<(Mpts * Cc) / 256, 256, 0, stream>>>(s, bnsc, bnsh);
  // SE-1: h = relu(s @ w1^T + b1)          (32768 x 64, KK=256)
  k_gemm_f32<<<dim3(Mpts / 64, 1), 256, 0, stream>>>(
      s, Cc, s, Cc, Cc, w1, Cc, b1, Cc, 1, nullptr, h, 64, 0);
  // SE-2: s = sigmoid(h @ w2^T + b2) * s   (in place, KK=64)
  k_gemm_f32<<<dim3(Mpts / 64, Cc / 64), 256, 0, stream>>>(
      h, 64, h, 64, 64, w2, 64, b2, 64, 2, s, s, Cc, 0);
  // Fusion: out = [x_t | s] @ wf^T + bf, stored transposed as (B,C,N)
  k_gemm_f32<<<dim3(Mpts / 64, Cc / 64), 256, 0, stream>>>(
      xt, Cc, s, Cc, Cc, wf, 2 * Cc, bf, 2 * Cc, 0, nullptr, out, Cc, 1);
}
